// SelfAttention_65910568125054
// MI455X (gfx1250) — matrix operand order repaired and checked by reading, not compiled, not run on hardware
//
#include <hip/hip_runtime.h>

typedef __bf16 v16bf __attribute__((ext_vector_type(16)));
typedef __bf16 v8bf  __attribute__((ext_vector_type(8)));
typedef float  v8f   __attribute__((ext_vector_type(8)));

#define B_ 8
#define T_ 2048
#define D_ 512

// ---------------------------------------------------------------------------
// Elementwise f32 -> bf16 (for the 512x512 weight matrices). 8 elems/thread.
// ---------------------------------------------------------------------------
__global__ __launch_bounds__(256) void cvtw_kernel(const float* __restrict__ in,
                                                   __bf16* __restrict__ out) {
  const int i = (blockIdx.x * 256 + threadIdx.x) * 8;
  const float4 f0 = *(const float4*)(in + i);
  const float4 f1 = *(const float4*)(in + i + 4);
  v8bf v;
  v[0] = (__bf16)f0.x; v[1] = (__bf16)f0.y; v[2] = (__bf16)f0.z; v[3] = (__bf16)f0.w;
  v[4] = (__bf16)f1.x; v[5] = (__bf16)f1.y; v[6] = (__bf16)f1.z; v[7] = (__bf16)f1.w;
  *(v8bf*)(out + i) = v;
}

// ---------------------------------------------------------------------------
// Projection: Out[r, e] = sum_d X[r, d] * W[e, d].  X: (B*T, D) f32,
// Wb: (D, D) bf16 [pre-converted]. Block = 8 waves, 16 rows x 512 cols,
// X tile staged once in LDS as bf16 (pitch 520 kills bank conflicts).
// TRANS=true writes V transposed (B, D, T). Grid: (B*T/16), block 256.
// ---------------------------------------------------------------------------
template <bool TRANS>
__global__ __launch_bounds__(256) void proj_kernel(const float* __restrict__ X,
                                                   const __bf16* __restrict__ Wb,
                                                   __bf16* __restrict__ Out) {
  __shared__ __bf16 x_lds[16][520];

  const int tid  = threadIdx.x;
  const int wave = tid >> 5;
  const int lane = tid & 31;
  const int half = (lane < 16) ? 0 : 1;
  const int l16  = lane & 15;

  const int rbase = blockIdx.x * 16;

  {
    const int r  = tid >> 4;
    const int c0 = (tid & 15) * 32;
    const float* xr = X + (size_t)(rbase + r) * D_ + c0;
#pragma unroll
    for (int u = 0; u < 4; ++u) {
      const float4 f0 = *(const float4*)(xr + u * 8);
      const float4 f1 = *(const float4*)(xr + u * 8 + 4);
      v8bf v;
      v[0] = (__bf16)f0.x; v[1] = (__bf16)f0.y; v[2] = (__bf16)f0.z; v[3] = (__bf16)f0.w;
      v[4] = (__bf16)f1.x; v[5] = (__bf16)f1.y; v[6] = (__bf16)f1.z; v[7] = (__bf16)f1.w;
      *(v8bf*)&x_lds[r][c0 + u * 8] = v;
    }
  }
  __syncthreads();

  v8f acc[4] = {};
  const int koff = half ? 16 : 0;
  for (int kk = 0; kk < D_; kk += 32) {
    v16bf a;
    {
      const v8bf alo = *(const v8bf*)&x_lds[l16][kk + koff];
      const v8bf ahi = *(const v8bf*)&x_lds[l16][kk + 8 + koff];
#pragma unroll
      for (int j = 0; j < 8; ++j) { a[j] = alo[j]; a[8 + j] = ahi[j]; }
    }
#pragma unroll
    for (int t = 0; t < 4; ++t) {
      const int nb = wave * 64 + t * 16;
      const v16bf bop = *(const v16bf*)(Wb + (size_t)(nb + l16) * D_ + kk + (half ? 16 : 0));
      acc[t] = __builtin_amdgcn_wmma_f32_16x16x32_bf16(false, a, false, bop,
                                                       (short)0, acc[t], false, false);
    }
  }

  if (TRANS) {
    const int bb   = rbase / T_;
    const int trow = rbase % T_;
#pragma unroll
    for (int t = 0; t < 4; ++t) {
      v8bf ov;
#pragma unroll
      for (int j = 0; j < 8; ++j) ov[j] = (__bf16)acc[t][j];
      *(v8bf*)(Out + ((size_t)bb * D_ + wave * 64 + t * 16 + l16) * T_ +
               trow + (half ? 8 : 0)) = ov;
    }
  } else {
#pragma unroll
    for (int t = 0; t < 4; ++t) {
#pragma unroll
      for (int j = 0; j < 8; ++j) {
        const int m = half ? (j + 8) : j;
        Out[(size_t)(rbase + m) * D_ + wave * 64 + t * 16 + l16] = (__bf16)acc[t][j];
      }
    }
  }
}

// ---------------------------------------------------------------------------
// Fused flash attention. Block = 8 waves, 32 queries x full D=512.
// Two 16-row query sub-tiles share the same K registers (2x K/V reuse).
// Waves split D for S; waves 0 and 1 each run the online softmax for one
// query sub-tile (mask read exactly once). P/corr double-buffered -> 2
// barriers per chunk; next-K prefetch is unconditional (wraps to 0 on the
// last iteration) to keep loads in flight across the barrier.
// Grid: (T/32, B).
// ---------------------------------------------------------------------------
__global__ __launch_bounds__(256) void attn_kernel(const __bf16* __restrict__ Qb,
                                                   const __bf16* __restrict__ Kb,
                                                   const __bf16* __restrict__ VbT,
                                                   const float* __restrict__ mask,
                                                   float* __restrict__ Out) {
  __shared__ float  s_red[8][2][2][32][8];  // [wave][qtile][ktile][lane][j]
  __shared__ __bf16 p_lds[2][32][32];       // [buf][qrow][kcol]
  __shared__ float  corr_lds[2][32];
  __shared__ float  inv_lds[32];

  const int tid  = threadIdx.x;
  const int wave = tid >> 5;
  const int lane = tid & 31;
  const int half = (lane < 16) ? 0 : 1;
  const int l16  = lane & 15;

  const int qbase = blockIdx.x * 32;
  const int bb    = blockIdx.y;
  const size_t bT = (size_t)bb * T_;
  const int koff  = half ? 16 : 0;

  // Q A-operands: qa[qtile][kslice] for D slice [wave*64, wave*64+64)
  v16bf qa[2][2];
#pragma unroll
  for (int q = 0; q < 2; ++q) {
    const __bf16* qr = Qb + (bT + qbase + q * 16 + l16) * D_ + wave * 64;
    const v8bf q0 = *(const v8bf*)(qr + koff);
    const v8bf q1 = *(const v8bf*)(qr + 8 + koff);
    const v8bf q2 = *(const v8bf*)(qr + 32 + koff);
    const v8bf q3 = *(const v8bf*)(qr + 40 + koff);
#pragma unroll
    for (int j = 0; j < 8; ++j) { qa[q][0][j] = q0[j]; qa[q][0][8 + j] = q1[j]; }
#pragma unroll
    for (int j = 0; j < 8; ++j) { qa[q][1][j] = q2[j]; qa[q][1][8 + j] = q3[j]; }
  }

  const __bf16* kp0 = Kb + (bT + l16) * D_ + wave * 64 + (half ? 16 : 0);
  const __bf16* kp1 = Kb + (bT + 16 + l16) * D_ + wave * 64 + (half ? 16 : 0);

  auto loadK = [&](int kb, v16bf* kr) {
    const size_t off = (size_t)kb * D_;
    kr[0] = *(const v16bf*)(kp0 + off);        // slice 0, keys 0..15
    kr[1] = *(const v16bf*)(kp1 + off);        // slice 0, keys 16..31
    kr[2] = *(const v16bf*)(kp0 + off + 32);   // slice 1, keys 0..15
    kr[3] = *(const v16bf*)(kp1 + off + 32);   // slice 1, keys 16..31
  };

  v8f o[2][4] = {};
  float rm[8], rl[8];  // live only in waves 0 / 1 (one qtile each)
#pragma unroll
  for (int j = 0; j < 8; ++j) { rm[j] = -1e30f; rl[j] = 0.0f; }

  const float SQ = 22.62741699796952f;  // sqrt(512)

  v16bf kr[4];
  loadK(0, kr);
  int buf = 0;

  for (int kbase = 0; kbase < T_; kbase += 32) {
    // ---- partial S (both query sub-tiles share the K registers)
#pragma unroll
    for (int q = 0; q < 2; ++q) {
      v8f s0 = {}, s1 = {};
      s0 = __builtin_amdgcn_wmma_f32_16x16x32_bf16(false, qa[q][0], false, kr[0], (short)0, s0, false, false);
      s1 = __builtin_amdgcn_wmma_f32_16x16x32_bf16(false, qa[q][0], false, kr[1], (short)0, s1, false, false);
      s0 = __builtin_amdgcn_wmma_f32_16x16x32_bf16(false, qa[q][1], false, kr[2], (short)0, s0, false, false);
      s1 = __builtin_amdgcn_wmma_f32_16x16x32_bf16(false, qa[q][1], false, kr[3], (short)0, s1, false, false);
      *(v8f*)&s_red[wave][q][0][lane][0] = s0;
      *(v8f*)&s_red[wave][q][1][lane][0] = s1;
    }

    // ---- prefetch: V tiles for THIS chunk, K operands for NEXT chunk
    v16bf vb[4];
#pragma unroll
    for (int t = 0; t < 4; ++t) {
      const int dcol = wave * 64 + t * 16 + l16;
      vb[t] = *(const v16bf*)(VbT + ((size_t)bb * D_ + dcol) * T_ +
                              kbase + (half ? 16 : 0));
    }
    const int knext = (kbase + 32 < T_) ? (kbase + 32) : 0;  // unconditional
    loadK(knext, kr);

    __syncthreads();

    // ---- waves 0,1: reduce + scale + mask + online softmax for own qtile
    if (wave < 2) {
      const int q = wave;
      v8f a0 = {}, a1 = {};
#pragma unroll
      for (int w = 0; w < 8; ++w) {
        a0 += *(const v8f*)&s_red[w][q][0][lane][0];
        a1 += *(const v8f*)&s_red[w][q][1][lane][0];
      }
      float f0[8], f1[8];
#pragma unroll
      for (int j = 0; j < 8; ++j) {
        const int qrow = qbase + q * 16 + (half ? j + 8 : j);
        const size_t mrow = (bT + qrow) * (size_t)T_ + kbase;
        f0[j] = a0[j] * SQ + mask[mrow + l16];
        f1[j] = a1[j] * SQ + mask[mrow + 16 + l16];
      }
#pragma unroll
      for (int j = 0; j < 8; ++j) {
        float mx = fmaxf(f0[j], f1[j]);
#pragma unroll
        for (int sh = 1; sh < 16; sh <<= 1) mx = fmaxf(mx, __shfl_xor(mx, sh, 32));
        const float nm = fmaxf(rm[j], mx);
        const float c  = __expf(rm[j] - nm);
        const float p0 = __expf(f0[j] - nm);
        const float p1 = __expf(f1[j] - nm);
        float rs = p0 + p1;
#pragma unroll
        for (int sh = 1; sh < 16; sh <<= 1) rs += __shfl_xor(rs, sh, 32);
        rl[j] = rl[j] * c + rs;
        rm[j] = nm;
        const int m = q * 16 + (half ? j + 8 : j);
        p_lds[buf][m][l16]      = (__bf16)p0;
        p_lds[buf][m][16 + l16] = (__bf16)p1;
        corr_lds[buf][m]        = c;
      }
    }
    __syncthreads();

    // ---- all waves: rescale O, read P per qtile, O += P @ V
#pragma unroll
    for (int q = 0; q < 2; ++q) {
#pragma unroll
      for (int j = 0; j < 8; ++j) {
        const float c = corr_lds[buf][q * 16 + (half ? j + 8 : j)];
        o[q][0][j] *= c; o[q][1][j] *= c; o[q][2][j] *= c; o[q][3][j] *= c;
      }
      v16bf pa;
      {
        const v8bf plo = *(const v8bf*)&p_lds[buf][q * 16 + l16][koff];
        const v8bf phi = *(const v8bf*)&p_lds[buf][q * 16 + l16][8 + koff];
#pragma unroll
        for (int j = 0; j < 8; ++j) { pa[j] = plo[j]; pa[8 + j] = phi[j]; }
      }
#pragma unroll
      for (int t = 0; t < 4; ++t) {
        o[q][t] = __builtin_amdgcn_wmma_f32_16x16x32_bf16(false, pa, false, vb[t],
                                                          (short)0, o[q][t], false, false);
      }
    }
    buf ^= 1;  // P/corr double-buffered: no end-of-loop barrier
  }

  // ---- epilogue
  if (wave < 2) {
#pragma unroll
    for (int j = 0; j < 8; ++j) inv_lds[wave * 16 + (half ? j + 8 : j)] = 1.0f / rl[j];
  }
  __syncthreads();
#pragma unroll
  for (int q = 0; q < 2; ++q) {
#pragma unroll
    for (int j = 0; j < 8; ++j) {
      const int m = q * 16 + (half ? j + 8 : j);
      const float inv = inv_lds[m];
      float* orow = Out + (bT + qbase + m) * D_ + wave * 64 + l16;
      orow[0]  = o[q][0][j] * inv;
      orow[16] = o[q][1][j] * inv;
      orow[32] = o[q][2][j] * inv;
      orow[48] = o[q][3][j] * inv;
    }
  }
}

// ---------------------------------------------------------------------------
extern "C" void kernel_launch(void* const* d_in, const int* in_sizes, int n_in,
                              void* d_out, int out_size, void* d_ws, size_t ws_size,
                              hipStream_t stream) {
  const float* ft_q = (const float*)d_in[0];
  const float* ft_k = (const float*)d_in[1];
  const float* ft_v = (const float*)d_in[2];
  const float* mask = (const float*)d_in[3];
  const float* Wq   = (const float*)d_in[4];
  const float* Wk   = (const float*)d_in[5];
  const float* Wv   = (const float*)d_in[6];
  float* out = (float*)d_out;

  const size_t MD = (size_t)B_ * T_ * D_;
  const size_t DD = (size_t)D_ * D_;
  __bf16* qb  = (__bf16*)d_ws;
  __bf16* kb  = qb + MD;
  __bf16* vbt = kb + MD;   // (B, D, T) transposed
  __bf16* wqb = vbt + MD;
  __bf16* wkb = wqb + DD;
  __bf16* wvb = wkb + DD;

  const int cgrid = (int)(DD / (256 * 8));
  cvtw_kernel<<<cgrid, 256, 0, stream>>>(Wq, wqb);
  cvtw_kernel<<<cgrid, 256, 0, stream>>>(Wk, wkb);
  cvtw_kernel<<<cgrid, 256, 0, stream>>>(Wv, wvb);

  dim3 pgrid(B_ * T_ / 16);
  proj_kernel<false><<<pgrid, 256, 0, stream>>>(ft_q, wqb, qb);
  proj_kernel<false><<<pgrid, 256, 0, stream>>>(ft_k, wkb, kb);
  proj_kernel<true ><<<pgrid, 256, 0, stream>>>(ft_v, wvb, vbt);

  dim3 agrid(T_ / 32, B_);
  attn_kernel<<<agrid, 256, 0, stream>>>(qb, kb, vbt, mask, out);
}
